// SubmanifoldSparseConvBlock_730144440361
// MI455X (gfx1250) — matrix operand order repaired and checked by reading, not compiled, not run on hardware
//
#include <hip/hip_runtime.h>
#include <stdint.h>

#define G 160
#define G3 (G * G * G)
#define CIN 32
#define COUT 64
#define NOFF 27
#define MT 4                 // M-subtiles (of 16 points) per wave -> 64 points/wave

typedef __attribute__((ext_vector_type(16))) __bf16 v16bf;
typedef __attribute__((ext_vector_type(8)))  float  v8f;
typedef __attribute__((ext_vector_type(4)))  int    v4i;

union V16 {
    v16bf v;
    uint4 q[2];
};

__device__ __forceinline__ unsigned short f2bf(float f) {
    union { float f; unsigned int u; } c;
    c.f = f;
    unsigned int u = c.u;
    unsigned int r = u + 0x7FFFu + ((u >> 16) & 1u);  // round-to-nearest-even
    return (unsigned short)(r >> 16);
}

// ---------------------------------------------------------------- init / prep
__global__ void k_init(int* __restrict__ table, float* __restrict__ stats, int g3) {
    int i = blockIdx.x * blockDim.x + threadIdx.x;
    if (i < g3) table[i] = -1;
    if (i < 128) stats[i] = 0.f;
}

__global__ void k_scatter(const int* __restrict__ pos, int* __restrict__ table, int n) {
    int i = blockIdx.x * blockDim.x + threadIdx.x;
    if (i >= n) return;
    int lin = (pos[3 * i] * G + pos[3 * i + 1]) * G + pos[3 * i + 2];
    table[lin] = i;
}

__global__ void k_cvt_feat(const float* __restrict__ f, unsigned short* __restrict__ h, int total) {
    int i = blockIdx.x * blockDim.x + threadIdx.x;
    if (i < total) h[i] = f2bf(f[i]);
}

// Pack W[27][32][64] (fp32) into the WMMA bf16 B-matrix VGPR layout:
// for (k, g, lane): 16 bf16 values, value j corresponds to K = 8*(lane>>4)+j (j<8), 16+8*(lane>>4)+(j-8) (j>=8),
// column co = g*16 + (lane&15).  32 bytes per lane, contiguous.
__global__ void k_pack_w(const float* __restrict__ W, unsigned short* __restrict__ wp) {
    int t = blockIdx.x * blockDim.x + threadIdx.x;
    if (t >= NOFF * 4 * 32) return;
    int lane = t & 31;
    int g    = (t >> 5) & 3;
    int k    = t >> 7;
    int n = lane & 15, h = lane >> 4;
    int co = g * 16 + n;
    unsigned short* dst = wp + (size_t)t * 16;
#pragma unroll
    for (int j = 0; j < 16; ++j) {
        int K = (j < 8) ? (8 * h + j) : (16 + 8 * h + (j - 8));
        dst[j] = f2bf(W[((size_t)k * CIN + K) * COUT + co]);
    }
}

// ---------------------------------------------------------------- main conv
__launch_bounds__(256)
__global__ void k_conv_wmma(const int* __restrict__ pos,
                            const int* __restrict__ table,
                            const unsigned short* __restrict__ featH,
                            const uint4* __restrict__ wpackG,
                            float* __restrict__ out,
                            float* __restrict__ stats,
                            int nsuper, int npts) {
    __shared__ uint4 sW[NOFF * 4 * 32 * 2];   // 27*4 B-tiles, 32 lanes * 32B = 110,592 B
    __shared__ float sred[256];               // [stat(2)][g(4)][lane(32)]

    const int tid = threadIdx.x;

    // Stage the packed W into LDS. Prefer the CDNA5 async global->LDS DMA path.
#if __has_builtin(__builtin_amdgcn_global_load_async_to_lds_b128)
    {
        typedef __attribute__((address_space(1))) v4i* g4p;
        typedef __attribute__((address_space(3))) v4i* l4p;
        for (int i = tid; i < NOFF * 4 * 32 * 2; i += 256)
            __builtin_amdgcn_global_load_async_to_lds_b128(
                (g4p)(wpackG + i), (l4p)(sW + i), 0, 0);
        asm volatile("s_wait_asynccnt 0x0" ::: "memory");
    }
#else
    for (int i = tid; i < NOFF * 4 * 32 * 2; i += 256) sW[i] = wpackG[i];
#endif
    sred[tid] = 0.f;
    __syncthreads();

    const int lane = tid & 31;
    const int n = lane & 15;
    const int h = lane >> 4;
    const int wave   = tid >> 5;
    const int gwave  = blockIdx.x * 8 + wave;
    const int nwaves = gridDim.x * 8;

    float lsum[4] = {0.f, 0.f, 0.f, 0.f};
    float lsq[4]  = {0.f, 0.f, 0.f, 0.f};
    const v8f vzero = {0.f, 0.f, 0.f, 0.f, 0.f, 0.f, 0.f, 0.f};

    for (int st = gwave; st < nsuper; st += nwaves) {
        const int base0 = st * (16 * MT);

        int px[MT], py[MT], pz[MT];
#pragma unroll
        for (int mt = 0; mt < MT; ++mt) {
            int p = base0 + mt * 16 + n;
            if (p >= npts) p = npts - 1;
            px[mt] = pos[3 * p];
            py[mt] = pos[3 * p + 1];
            pz[mt] = pos[3 * p + 2];
        }

        v8f acc[MT][4];
#pragma unroll
        for (int mt = 0; mt < MT; ++mt)
#pragma unroll
            for (int g = 0; g < 4; ++g) acc[mt][g] = vzero;

#pragma unroll 1
        for (int k = 0; k < NOFF; ++k) {
            const int dx = k / 9 - 1, dy = (k / 3) % 3 - 1, dz = k % 3 - 1;

            // Gather A-tiles for all MT subtiles. 16-bit A 16x32 layout:
            // lane(h=0) takes K 0..7 & 16..23, lane(h=1) takes K 8..15 & 24..31.
            V16 a[MT];
#pragma unroll
            for (int mt = 0; mt < MT; ++mt) {
                const int nx = px[mt] + dx, ny = py[mt] + dy, nz = pz[mt] + dz;
                int idx = -1;
                if ((unsigned)nx < (unsigned)G && (unsigned)ny < (unsigned)G &&
                    (unsigned)nz < (unsigned)G)
                    idx = table[(nx * G + ny) * G + nz];
                if (idx >= 0) {
                    const uint4* rp = (const uint4*)(featH + (size_t)idx * CIN);
                    a[mt].q[0] = rp[h];
                    a[mt].q[1] = rp[h + 2];
                } else {
                    a[mt].q[0] = make_uint4(0u, 0u, 0u, 0u);
                    a[mt].q[1] = make_uint4(0u, 0u, 0u, 0u);
                }
            }

            // Each B tile read once from LDS feeds MT WMMAs.
            const int bbase = k * 4 * 64;  // uint4 index; (k,g) tile = 64 uint4
#pragma unroll
            for (int g = 0; g < 4; ++g) {
                V16 b;
                b.q[0] = sW[bbase + g * 64 + lane * 2 + 0];
                b.q[1] = sW[bbase + g * 64 + lane * 2 + 1];
#pragma unroll
                for (int mt = 0; mt < MT; ++mt)
                    acc[mt][g] = __builtin_amdgcn_wmma_f32_16x16x32_bf16(
                        false, a[mt].v, false, b.v, (short)0, acc[mt][g], false, false);
            }
        }

        // Store conv output (fp32) and accumulate per-channel BN partials.
        // C/D layout: VGPR r -> point m = r + 8h, lane -> channel co = 16g + n.
#pragma unroll
        for (int mt = 0; mt < MT; ++mt) {
            const int base = base0 + mt * 16;
#pragma unroll
            for (int g = 0; g < 4; ++g) {
                const int co = g * 16 + n;
                float s = 0.f, s2 = 0.f;
#pragma unroll
                for (int r = 0; r < 8; ++r) {
                    float x = acc[mt][g][r];
                    int m = base + r + 8 * h;
                    if (m < npts) out[(size_t)m * COUT + co] = x;
                    s += x;
                    s2 += x * x;
                }
                lsum[g] += s;
                lsq[g]  += s2;
            }
        }
    }

    // workgroup reduction of BN partials via LDS atomics, then global atomics
#pragma unroll
    for (int g = 0; g < 4; ++g) {
        atomicAdd(&sred[0 * 128 + g * 32 + lane], lsum[g]);
        atomicAdd(&sred[1 * 128 + g * 32 + lane], lsq[g]);
    }
    __syncthreads();
    if (tid < 128) {
        const int stat = tid >> 6;
        const int ch = tid & 63;
        const int g = ch >> 4, nn = ch & 15;
        float v = sred[stat * 128 + g * 32 + nn] + sred[stat * 128 + g * 32 + nn + 16];
        atomicAdd(&stats[stat * 64 + ch], v);
    }
}

// ---------------------------------------------------------------- BN + ReLU
__global__ void k_bn_relu(float* __restrict__ x, const float* __restrict__ stats,
                          const float* __restrict__ gamma, const float* __restrict__ beta,
                          int npts) {
    long long i = (long long)blockIdx.x * blockDim.x + threadIdx.x;
    long long total = (long long)npts * COUT;
    if (i >= total) return;
    int c = (int)(i & (COUT - 1));
    float invn = 1.0f / (float)npts;
    float mean = stats[c] * invn;
    float var  = stats[64 + c] * invn - mean * mean;
    float inv  = rsqrtf(var + 1e-5f);
    float y = gamma[c] * (x[i] - mean) * inv + beta[c];
    x[i] = y > 0.f ? y : 0.f;
}

__global__ void k_copy_pos(const int* __restrict__ pos, int* __restrict__ dst, int total) {
    int i = blockIdx.x * blockDim.x + threadIdx.x;
    if (i < total) dst[i] = pos[i];
}

// ---------------------------------------------------------------- launcher
extern "C" void kernel_launch(void* const* d_in, const int* in_sizes, int n_in,
                              void* d_out, int out_size, void* d_ws, size_t ws_size,
                              hipStream_t stream) {
    const float* features  = (const float*)d_in[0];
    const int*   positions = (const int*)d_in[1];
    const float* W         = (const float*)d_in[2];
    const float* gamma     = (const float*)d_in[3];
    const float* beta      = (const float*)d_in[4];
    const int N = in_sizes[1] / 3;

    // workspace layout
    char* ws = (char*)d_ws;
    int* table = (int*)ws;                                   // 16.4 MB
    size_t off = (size_t)G3 * 4;
    unsigned short* featH = (unsigned short*)(ws + off);     // 64 MB (bf16)
    off += (size_t)N * CIN * 2;
    off = (off + 255) & ~(size_t)255;
    unsigned short* wpack = (unsigned short*)(ws + off);     // 108 KB
    off += (size_t)NOFF * 4 * 32 * 16 * 2;
    off = (off + 255) & ~(size_t)255;
    float* stats = (float*)(ws + off);                       // 512 B

    float* convout = (float*)d_out;
    int*   posout  = (int*)((float*)d_out + (size_t)N * COUT);

    k_init<<<(G3 + 255) / 256, 256, 0, stream>>>(table, stats, G3);
    k_scatter<<<(N + 255) / 256, 256, 0, stream>>>(positions, table, N);
    k_cvt_feat<<<(N * CIN + 255) / 256, 256, 0, stream>>>(features, featH, N * CIN);
    k_pack_w<<<(NOFF * 4 * 32 + 255) / 256, 256, 0, stream>>>(W, wpack);

    const int nsuper = (N + 16 * MT - 1) / (16 * MT);
    k_conv_wmma<<<512, 256, 0, stream>>>(positions, table, featH,
                                         (const uint4*)wpack, convout, stats,
                                         nsuper, N);

    long long total = (long long)N * COUT;
    k_bn_relu<<<(int)((total + 255) / 256), 256, 0, stream>>>(convout, stats, gamma, beta, N);
    k_copy_pos<<<(N * 3 + 255) / 256, 256, 0, stream>>>(positions, posout, N * 3);
}
